// DistributionShiftGAT_76922864271378
// MI455X (gfx1250) — matrix operand order repaired and checked by reading, not compiled, not run on hardware
//
#include <hip/hip_runtime.h>
#include <cstdint>
#include <cstddef>

// ---------------------------------------------------------------------------
// GAT 3-layer pipeline for gfx1250 (MI455X).
// GEMMs run on the matrix pipe via v_wmma_f32_16x16x32_bf16 (bf16 in, f32 acc),
// with compile-time N/K, b128 global->LDS tile staging, global_prefetch of the
// next K tile, and a branch-free epilogue fast path.
// Edge softmax/aggregation via ordered-int atomicMax + float atomic adds.
// ---------------------------------------------------------------------------

typedef __attribute__((ext_vector_type(16))) __bf16 v16bf;
typedef __attribute__((ext_vector_type(8)))  float  v8f;

#define LDSS 40   // LDS row stride in bf16 elems (80 B, 16-B aligned for b128)

// ---- small helpers --------------------------------------------------------

__device__ __forceinline__ uint16_t f32_to_bf16(float f) {
  uint32_t u = __float_as_uint(f);
  u += 0x7fffu + ((u >> 16) & 1u);   // round-to-nearest-even
  return (uint16_t)(u >> 16);
}

// monotone float <-> signed-int encoding (self-inverse) for atomicMax on floats
__device__ __forceinline__ int enc_f(float f) {
  int i = __float_as_int(f);
  return i ^ ((i >> 31) & 0x7fffffff);
}
__device__ __forceinline__ float dec_i(int i) {
  int j = i ^ ((i >> 31) & 0x7fffffff);
  return __int_as_float(j);
}

__device__ __forceinline__ void atomAddF(float* p, float v) {
#if __has_builtin(__builtin_amdgcn_global_atomic_fadd_f32)
  __builtin_amdgcn_global_atomic_fadd_f32(p, v);
#else
  atomicAdd(p, v);
#endif
}

// ---- utility kernels ------------------------------------------------------

__global__ void k_zero_f32(float* p, int n) {
  int t = blockIdx.x * blockDim.x + threadIdx.x;
  if (t < n) p[t] = 0.0f;
}

__global__ void k_fill_i32(int* p, int n, int v) {
  int t = blockIdx.x * blockDim.x + threadIdx.x;
  if (t < n) p[t] = v;
}

__global__ void k_cvt_bf16(const float* __restrict__ in, uint16_t* __restrict__ out, int n) {
  int t = blockIdx.x * blockDim.x + threadIdx.x;
  if (t < n) out[t] = f32_to_bf16(in[t]);
}

// W [K,N] f32 (row-major)  ->  Wt [N,K] bf16 (row-major). Small matrices only.
__global__ void k_cvt_transpose_bf16(const float* __restrict__ in,
                                     uint16_t* __restrict__ out, int K, int N) {
  int t = blockIdx.x * blockDim.x + threadIdx.x;
  if (t >= K * N) return;
  int k = t / N, n = t - k * N;
  out[(size_t)n * K + k] = f32_to_bf16(in[t]);
}

// ---- tiled bf16 WMMA GEMM:  C[M,N] = A[M,K] * Bt[N,K]^T -------------------
// Block: 256 threads (8 waves). Block tile 128x128, wave tile 32x64,
// K stepped by 32 -> 8 x v_wmma_f32_16x16x32_bf16 per wave per K-step.
// N, K compile-time so epilogue strides fold into immediate offsets.

template <int K, int N>
__global__ __launch_bounds__(256) void k_gemm_bf16_wmma(
    const uint16_t* __restrict__ A, const uint16_t* __restrict__ Bt,
    float* __restrict__ C, int M)
{
  __shared__ __align__(16) uint16_t As[128 * LDSS];   // [m][k]
  __shared__ __align__(16) uint16_t Bs[128 * LDSS];   // [n][k]

  const int tid  = threadIdx.x;
  const int lane = tid & 31;
  const int wave = tid >> 5;
  const int waveM = wave & 3;           // 4 waves along M (32 rows each)
  const int waveN = wave >> 2;          // 2 waves along N (64 cols each)
  const int blockM = blockIdx.y * 128;
  const int blockN = blockIdx.x * 128;
  const int g   = lane >> 4;            // lane group 0/1
  const int l16 = lane & 15;

  // tile-staging assignments (each thread: 32 B of A + 32 B of B per tile)
  const int r  = tid >> 1;              // row within tile (0..127)
  const int ks = (tid & 1) * 16;        // k start (0 or 16)
  int gr = blockM + r; if (gr >= M) gr = M - 1;
  const uint16_t* aGlob = A  + (size_t)gr * K + ks;            // + k0 later
  const uint16_t* bGlob = Bt + (size_t)(blockN + r) * K + ks;  // + k0 later

  v8f acc[2][4];
  #pragma unroll
  for (int i = 0; i < 2; i++)
    #pragma unroll
    for (int j = 0; j < 4; j++)
      #pragma unroll
      for (int e = 0; e < 8; e++) acc[i][j][e] = 0.0f;

  for (int k0 = 0; k0 < K; k0 += 32) {
    __syncthreads();
    {
      const uint4* as = (const uint4*)(aGlob + k0);
      uint4* ad = (uint4*)&As[r * LDSS + ks];
      ad[0] = as[0]; ad[1] = as[1];
      const uint4* bs = (const uint4*)(bGlob + k0);
      uint4* bd = (uint4*)&Bs[r * LDSS + ks];
      bd[0] = bs[0]; bd[1] = bs[1];
      if (k0 + 32 < K) {   // prefetch next K tile (global_prefetch_b8)
        __builtin_prefetch(aGlob + k0 + 32, 0, 1);
        __builtin_prefetch(bGlob + k0 + 32, 0, 1);
      }
    }
    __syncthreads();

    // Build WMMA fragments per ISA 7.12.2 (16-bit A 16x32 / B 32x16 layouts)
    union Frag { uint32_t u[8]; v16bf v; };
    Frag a[2], b[4];
    #pragma unroll
    for (int tm = 0; tm < 2; tm++) {
      int Mr = waveM * 32 + tm * 16 + l16;
      #pragma unroll
      for (int v = 0; v < 8; v++) {
        int kp = (v >> 2) * 16 + g * 8 + (v & 3) * 2;   // K pair base
        a[tm].u[v] = *(const uint32_t*)&As[Mr * LDSS + kp];
      }
    }
    #pragma unroll
    for (int tn = 0; tn < 4; tn++) {
      int Nc = waveN * 64 + tn * 16 + l16;
      #pragma unroll
      for (int v = 0; v < 8; v++) {
        int kp = (v >> 2) * 16 + g * 8 + (v & 3) * 2;
        b[tn].u[v] = *(const uint32_t*)&Bs[Nc * LDSS + kp];
      }
    }

    #pragma unroll
    for (int tm = 0; tm < 2; tm++)
      #pragma unroll
      for (int tn = 0; tn < 4; tn++)
        acc[tm][tn] = __builtin_amdgcn_wmma_f32_16x16x32_bf16(
            false, a[tm].v, false, b[tn].v, (short)0, acc[tm][tn], false, false);
  }

  // Store: C/D layout -> lane l: N = l%16, VGPR v: M = v + (l/16)*8
  const int colBase = blockN + waveN * 64 + l16;
  if (blockM + 128 <= M) {
    // fast path: whole block in range, branch-free stores w/ immediate offsets
    #pragma unroll
    for (int tm = 0; tm < 2; tm++) {
      float* p = C + (size_t)(blockM + waveM * 32 + tm * 16 + g * 8) * N + colBase;
      #pragma unroll
      for (int tn = 0; tn < 4; tn++)
        #pragma unroll
        for (int v = 0; v < 8; v++)
          p[(size_t)v * N + tn * 16] = acc[tm][tn][v];
    }
  } else {
    #pragma unroll
    for (int tm = 0; tm < 2; tm++) {
      int Mr0 = blockM + waveM * 32 + tm * 16 + g * 8;
      float* p = C + (size_t)Mr0 * N + colBase;
      #pragma unroll
      for (int tn = 0; tn < 4; tn++)
        #pragma unroll
        for (int v = 0; v < 8; v++)
          if (Mr0 + v < M) p[(size_t)v * N + tn * 16] = acc[tm][tn][v];
    }
  }
}

// ---- attention scores: out[n*H+h] = dot(h[n,h,:], att[h,:]) ---------------
// one wave32 per (node, head)

__global__ void k_attn_scores(const float* __restrict__ h, const float* __restrict__ att,
                              float* __restrict__ out, int Nn, int H, int C)
{
  int pair = (blockIdx.x * blockDim.x + threadIdx.x) >> 5;
  int lane = threadIdx.x & 31;
  if (pair >= Nn * H) return;
  int n = pair / H, hd = pair - n * H;
  const float* hp = h + (size_t)n * H * C + hd * C;
  const float* ap = att + hd * C;
  float s = 0.0f;
  for (int c = lane; c < C; c += 32) s += hp[c] * ap[c];
  #pragma unroll
  for (int off = 16; off; off >>= 1) s += __shfl_xor(s, off, 32);
  if (lane == 0) out[pair] = s;
}

// ---- edge pass 1: e = leaky_relu(a_src[src]+a_dst[dst]); segment max ------

__global__ void k_edge_scores(const long long* __restrict__ ei, int E, int Etot, int H,
                              const float* __restrict__ asrc, const float* __restrict__ adst,
                              float* __restrict__ ebuf, int* __restrict__ m)
{
  int t = blockIdx.x * blockDim.x + threadIdx.x;
  if (t >= Etot * H) return;
  int e = t / H, hd = t - e * H;
  int s, d;
  if (e < E) { s = (int)ei[e]; d = (int)ei[E + e]; } else { s = d = e - E; }
  float v = asrc[s * H + hd] + adst[d * H + hd];
  v = v > 0.0f ? v : 0.2f * v;       // leaky_relu, slope 0.2
  ebuf[t] = v;
  atomicMax(&m[d * H + hd], enc_f(v));
}

// decode ordered-int max -> float in place; non-finite -> 0
__global__ void k_finalize_max(int* m, int n) {
  int t = blockIdx.x * blockDim.x + threadIdx.x;
  if (t >= n) return;
  float f = dec_i(m[t]);
  if (!isfinite(f)) f = 0.0f;
  ((float*)m)[t] = f;
}

// ---- edge pass 2: ex = exp(e - m[dst]); segment sum -----------------------

__global__ void k_edge_exp(const long long* __restrict__ ei, int E, int Etot, int H,
                           const float* __restrict__ mf, float* __restrict__ ebuf,
                           float* __restrict__ denom)
{
  int t = blockIdx.x * blockDim.x + threadIdx.x;
  if (t >= Etot * H) return;
  int e = t / H, hd = t - e * H;
  int d = (e < E) ? (int)ei[E + e] : e - E;
  float ex = __expf(ebuf[t] - mf[d * H + hd]);
  ebuf[t] = ex;
  atomAddF(&denom[d * H + hd], ex);
}

// ---- edge pass 3: out[dst] += (ex/denom[dst]) * h[src] --------------------
// one thread per (edge, head, 4-channel chunk)

__global__ void k_edge_aggregate(const long long* __restrict__ ei, int E, int Etot,
                                 int H, int C,
                                 const float* __restrict__ ebuf,
                                 const float* __restrict__ denom,
                                 const float* __restrict__ h,
                                 float* __restrict__ out)
{
  int C4 = C >> 2;
  long long t = (long long)blockIdx.x * blockDim.x + threadIdx.x;
  long long total = (long long)Etot * H * C4;
  if (t >= total) return;
  int c4 = (int)(t % C4);
  long long eh = t / C4;
  int hd = (int)(eh % H);
  int e  = (int)(eh / H);
  int s, d;
  if (e < E) { s = (int)ei[e]; d = (int)ei[E + e]; } else { s = d = e - E; }
  float alpha = ebuf[(size_t)e * H + hd] / denom[d * H + hd];
  const float4 hv = *(const float4*)&h[(size_t)s * H * C + hd * C + c4 * 4];
  float* op = &out[(size_t)d * H * C + hd * C + c4 * 4];
  atomAddF(op + 0, alpha * hv.x);
  atomAddF(op + 1, alpha * hv.y);
  atomAddF(op + 2, alpha * hv.z);
  atomAddF(op + 3, alpha * hv.w);
}

// ---- bias (+ optional relu), in place -------------------------------------

__global__ void k_bias_act(float* x, const float* __restrict__ b, int n, int F, int relu) {
  int t = blockIdx.x * blockDim.x + threadIdx.x;
  if (t >= n) return;
  float v = x[t] + b[t % F];
  x[t] = relu ? fmaxf(v, 0.0f) : v;
}

// ---------------------------------------------------------------------------

extern "C" void kernel_launch(void* const* d_in, const int* in_sizes, int n_in,
                              void* d_out, int out_size, void* d_ws, size_t ws_size,
                              hipStream_t stream)
{
  (void)n_in; (void)out_size; (void)ws_size;

  const float*     x   = (const float*)d_in[0];
  const long long* ei  = (const long long*)d_in[1];   // int64 edge_index [2,E]
  const float* W1  = (const float*)d_in[2];
  const float* as1 = (const float*)d_in[3];
  const float* ad1 = (const float*)d_in[4];
  const float* b1  = (const float*)d_in[5];
  const float* W2  = (const float*)d_in[6];
  const float* as2 = (const float*)d_in[7];
  const float* ad2 = (const float*)d_in[8];
  const float* b2  = (const float*)d_in[9];
  const float* W3  = (const float*)d_in[10];
  const float* as3 = (const float*)d_in[11];
  const float* ad3 = (const float*)d_in[12];
  const float* b3  = (const float*)d_in[13];
  float* outp = (float*)d_out;

  const int Nn   = in_sizes[0] / 256;   // 30000
  const int Fin  = 256;
  const int HC   = 512;                 // 4 heads * 128
  const int E    = in_sizes[1] / 2;     // 480000
  const int Etot = E + Nn;              // + self loops

  // ---- carve workspace ----
  size_t off = 0;
  auto carve = [&](size_t bytes) {
    size_t o = off; off += (bytes + 255) & ~(size_t)255; return (char*)d_ws + o;
  };
  float*    hbuf  = (float*)   carve((size_t)Nn * HC * 4);   // GEMM output h
  float*    obuf  = (float*)   carve((size_t)Nn * HC * 4);   // aggregated / layer input
  uint16_t* abuf  = (uint16_t*)carve((size_t)Nn * HC * 2);   // bf16 activations
  uint16_t* wbuf  = (uint16_t*)carve((size_t)HC * HC * 2);   // bf16 W^T [N,K]
  float*    asrcb = (float*)   carve((size_t)Nn * 4 * 4);
  float*    adstb = (float*)   carve((size_t)Nn * 4 * 4);
  int*      mbuf  = (int*)     carve((size_t)Nn * 4 * 4);
  float*    dbuf  = (float*)   carve((size_t)Nn * 4 * 4);
  float*    ebuf  = (float*)   carve((size_t)Etot * 4 * 4);

  const int TB = 256;
  auto blks = [&](long long n) { return (unsigned)((n + TB - 1) / TB); };
  const int NEG_INF_ENC = (int)0x807FFFFF;  // enc_f(-inf)

  // one GAT layer edge phase: h [Nn, H*C] -> dst [Nn, H*C] (aggregated)
  auto gat_edge_phase = [&](const float* hfeat, const float* att_s, const float* att_d,
                            int H, int C, float* dst) {
    int NH = Nn * H;
    k_fill_i32<<<blks(NH), TB, 0, stream>>>(mbuf, NH, NEG_INF_ENC);
    k_zero_f32<<<blks(NH), TB, 0, stream>>>(dbuf, NH);
    k_zero_f32<<<blks((long long)Nn * H * C), TB, 0, stream>>>(dst, Nn * H * C);
    k_attn_scores<<<blks((long long)NH * 32), TB, 0, stream>>>(hfeat, att_s, asrcb, Nn, H, C);
    k_attn_scores<<<blks((long long)NH * 32), TB, 0, stream>>>(hfeat, att_d, adstb, Nn, H, C);
    k_edge_scores<<<blks((long long)Etot * H), TB, 0, stream>>>(ei, E, Etot, H, asrcb, adstb, ebuf, mbuf);
    k_finalize_max<<<blks(NH), TB, 0, stream>>>(mbuf, NH);
    k_edge_exp<<<blks((long long)Etot * H), TB, 0, stream>>>(ei, E, Etot, H, (float*)mbuf, ebuf, dbuf);
    k_edge_aggregate<<<blks((long long)Etot * H * (C >> 2)), TB, 0, stream>>>(
        ei, E, Etot, H, C, ebuf, dbuf, hfeat, dst);
  };

  // ================= Layer 1: [Nn,256] -> [Nn,512], 4 heads =================
  k_cvt_bf16<<<blks((long long)Nn * Fin), TB, 0, stream>>>(x, abuf, Nn * Fin);
  k_cvt_transpose_bf16<<<blks((long long)Fin * HC), TB, 0, stream>>>(W1, wbuf, Fin, HC);
  {
    dim3 grid(HC / 128, (Nn + 127) / 128);
    k_gemm_bf16_wmma<256, 512><<<grid, 256, 0, stream>>>(abuf, wbuf, hbuf, Nn);
  }
  gat_edge_phase(hbuf, as1, ad1, 4, 128, obuf);
  k_bias_act<<<blks((long long)Nn * HC), TB, 0, stream>>>(obuf, b1, Nn * HC, HC, 1);

  // ================= Layer 2: [Nn,512] -> [Nn,512], 4 heads =================
  k_cvt_bf16<<<blks((long long)Nn * HC), TB, 0, stream>>>(obuf, abuf, Nn * HC);
  k_cvt_transpose_bf16<<<blks((long long)HC * HC), TB, 0, stream>>>(W2, wbuf, HC, HC);
  {
    dim3 grid(HC / 128, (Nn + 127) / 128);
    k_gemm_bf16_wmma<512, 512><<<grid, 256, 0, stream>>>(abuf, wbuf, hbuf, Nn);
  }
  gat_edge_phase(hbuf, as2, ad2, 4, 128, obuf);
  k_bias_act<<<blks((long long)Nn * HC), TB, 0, stream>>>(obuf, b2, Nn * HC, HC, 1);

  // ================= Layer 3: [Nn,512] -> [Nn,128], 1 head (mean == id) =====
  k_cvt_bf16<<<blks((long long)Nn * HC), TB, 0, stream>>>(obuf, abuf, Nn * HC);
  k_cvt_transpose_bf16<<<blks((long long)HC * 128), TB, 0, stream>>>(W3, wbuf, HC, 128);
  {
    dim3 grid(1, (Nn + 127) / 128);
    k_gemm_bf16_wmma<512, 128><<<grid, 256, 0, stream>>>(abuf, wbuf, hbuf, Nn);
  }
  gat_edge_phase(hbuf, as3, ad3, 1, 128, outp);
  k_bias_act<<<blks((long long)Nn * 128), TB, 0, stream>>>(outp, b3, Nn * 128, 128, 0);
}
